// VanillaRNN_78305843741046
// MI455X (gfx1250) — matrix operand order repaired and checked by reading, not compiled, not run on hardware
//
#include <hip/hip_runtime.h>
#include <hip/hip_bf16.h>
#include <hip/hip_fp16.h>

// ---------------------------------------------------------------------------
// 2-layer vanilla RNN LM on gfx1250 (MI455X).
// Matmuls: v_wmma_f32_16x16x32_f16 (f16 operands, f32 accumulate).
// Tile staging: global_load_async_to_lds_b128 (ASYNCcnt), double-buffered LDS.
// ---------------------------------------------------------------------------

typedef _Float16 f16;
typedef __attribute__((ext_vector_type(16))) _Float16 v16h;
typedef __attribute__((ext_vector_type(8)))  _Float16 h8;
typedef __attribute__((ext_vector_type(8)))  float    v8f;

constexpr int B = 64, T = 1024, V = 512, E = 512, H = 1024;
constexpr int KT  = 32;   // K-tile (one WMMA K step)
constexpr int LDT = 40;   // LDS row stride in halves (80B: 16B-aligned, bank-spread)

// ---------------------------------------------------------------------------
// WMMA wrapper: D = A(16x32 f16) * B(32x16 f16) + C(16x16 f32)
// ---------------------------------------------------------------------------
__device__ __forceinline__ v8f wmma_16x16x32(v16h a, v16h b, v8f c) {
  return __builtin_amdgcn_wmma_f32_16x16x32_f16(
      /*neg_a=*/false, a, /*neg_b=*/false, b,
      /*c_mod=*/(short)0, c, /*reuse_a=*/false, /*reuse_b=*/false);
}

// A fragment (16x32, M rows x K cols), per ISA 7.12.2:
//   lane (mr=lane&15, hi=lane>>4) holds row M0+mr,
//   halves [0..7]  = K in [hi*8,    hi*8+8)
//   halves [8..15] = K in [16+hi*8, 16+hi*8+8)
__device__ __forceinline__ v16h frag_a(const f16 As[][LDT], int M0, int lane) {
  int mr = lane & 15, hi = lane >> 4;
  union { v16h v; h8 h[2]; } u;
  u.h[0] = *(const h8*)&As[M0 + mr][hi * 8];
  u.h[1] = *(const h8*)&As[M0 + mr][16 + hi * 8];
  return u.v;
}

// B fragment (32x16, K rows x N cols), stored in LDS transposed as Ws[N][K]
// (== weight rows directly, since out = act @ W^T with W row-major [N,K]):
//   lane (nr=lane&15, hi=lane>>4) holds col N0+nr,
//   halves [0..7] = K in [hi*8, hi*8+8), halves [8..15] = K in [16+hi*8, 16+hi*8+8)
__device__ __forceinline__ v16h frag_b(const f16 Ws[][LDT], int N0, int lane) {
  int nr = lane & 15, hi = lane >> 4;
  union { v16h v; h8 h[2]; } u;
  u.h[0] = *(const h8*)&Ws[N0 + nr][hi * 8];
  u.h[1] = *(const h8*)&Ws[N0 + nr][16 + hi * 8];
  return u.v;
}

// ---------------------------------------------------------------------------
// Async global -> LDS 16-byte copy (CDNA5, tracked by ASYNCcnt).
// LDS operand is the low 32 bits of the generic shared pointer (= AS3 offset,
// same value the compiler feeds ds_* ops); global operand is a VGPR pair.
// ---------------------------------------------------------------------------
__device__ __forceinline__ void async_b128(const f16* g, f16* l) {
  unsigned long long ga = (unsigned long long)(uintptr_t)g;
  unsigned int la = (unsigned int)(uintptr_t)l;
  asm volatile("global_load_async_to_lds_b128 %0, %1, off"
               :: "v"(la), "v"(ga) : "memory");
}
__device__ __forceinline__ void wait_async_le2() {
  asm volatile("s_wait_asynccnt 0x2" ::: "memory");
}
__device__ __forceinline__ void wait_async_0() {
  asm volatile("s_wait_asynccnt 0x0" ::: "memory");
}

// ---------------------------------------------------------------------------
// Double-buffered async K-loop shared by all GEMMs.
// aChunk / wChunk: this thread's first 16B source chunk; advances KT halves
// per iteration (row strides already folded in by the caller).
// Per iteration each thread issues 2 async ops; in-order retirement means
// "asynccnt <= 2" completes the current buffer while the next stays in flight.
// ---------------------------------------------------------------------------
__device__ __forceinline__ void gemm_k_loop(
    f16 (*As)[64][LDT], f16 (*Ws)[64][LDT],
    const f16* aChunk, const f16* wChunk, int nIter,
    int tid, int lane, int N0, int M0a, int M0b, v8f& c0, v8f& c1) {
  int row = tid >> 2, c = tid & 3;
  f16* aDst0 = &As[0][row][c * 8];
  f16* aDst1 = &As[1][row][c * 8];
  f16* wDst0 = &Ws[0][row][c * 8];
  f16* wDst1 = &Ws[1][row][c * 8];

  async_b128(aChunk, aDst0);
  async_b128(wChunk, wDst0);

  for (int i = 0; i < nIter; ++i) {
    int cur = i & 1;
    if (i + 1 < nIter) {
      const f16* aNext = aChunk + (size_t)(i + 1) * KT;
      const f16* wNext = wChunk + (size_t)(i + 1) * KT;
      async_b128(aNext, cur ? aDst0 : aDst1);
      async_b128(wNext, cur ? wDst0 : wDst1);
      wait_async_le2();   // current buffer landed; next 2 ops still in flight
    } else {
      wait_async_0();     // nothing else in flight: drain fully
    }
    __syncthreads();      // all waves' tile data visible in LDS

    v16h bfrag = frag_b(Ws[cur], N0, lane);
    c0 = wmma_16x16x32(frag_a(As[cur], M0a, lane), bfrag, c0);
    c1 = wmma_16x16x32(frag_a(As[cur], M0b, lane), bfrag, c1);

    __syncthreads();      // everyone done reading buf[cur] before overwrite
  }
}

// ---------------------------------------------------------------------------
// Small utility kernels
// ---------------------------------------------------------------------------
__global__ void cvt_f32_to_f16(const float* __restrict__ s, f16* __restrict__ d, int n) {
  for (int i = blockIdx.x * blockDim.x + threadIdx.x; i < n; i += gridDim.x * blockDim.x)
    d[i] = (f16)s[i];
}

__global__ void zero_f16(f16* __restrict__ d, int n) {
  for (int i = blockIdx.x * blockDim.x + threadIdx.x; i < n; i += gridDim.x * blockDim.x)
    d[i] = (f16)0.0f;
}

// Final hidden states -> f32 tail of d_out  (layout: [logits | h0 | h1])
__global__ void tail_states(const f16* __restrict__ h0f, const f16* __restrict__ h1f,
                            float* __restrict__ out) {
  size_t base = (size_t)B * T * V;
  int i = blockIdx.x * blockDim.x + threadIdx.x;
  if (i < B * H) {
    out[base + i]         = (float)h0f[i];
    out[base + B * H + i] = (float)h1f[i];
  }
}

// ---------------------------------------------------------------------------
// GEMM 1 (parallel): A0[t*B+b][n] = emb[x[b,t]] . W_ih0[n] + b_ih0[n]  (f16 out)
// grid = (H/64, B*T/64), block = 256
// ---------------------------------------------------------------------------
__global__ __launch_bounds__(256)
void embed_input_gemm(const int* __restrict__ x, const f16* __restrict__ emb,
                      const f16* __restrict__ Wih0, const float* __restrict__ bih0,
                      f16* __restrict__ A0) {
  __shared__ f16 As[2][64][LDT];
  __shared__ f16 Ws[2][64][LDT];
  int tid = threadIdx.x, lane = tid & 31, w = tid >> 5;
  int N0 = (w >> 1) * 16, M0a = (w & 1) * 32, M0b = M0a + 16;
  int mrow0 = blockIdx.y * 64, ncol0 = blockIdx.x * 64;
  v8f c0 = {}, c1 = {};

  // gather: GEMM row m = t*B + b  ->  b = m&63, t = m>>6
  int row = tid >> 2, cc = tid & 3;
  int m = mrow0 + row;
  int tok = x[(m & 63) * T + (m >> 6)];
  const f16* aChunk = emb + (size_t)tok * E + cc * 8;
  const f16* wChunk = Wih0 + (size_t)(ncol0 + row) * E + cc * 8;

  gemm_k_loop(As, Ws, aChunk, wChunk, E / KT, tid, lane, N0, M0a, M0b, c0, c1);

  int mr = lane & 15, hi = lane >> 4;
  int col = ncol0 + N0 + mr;
  float bias = bih0[col];
#pragma unroll
  for (int r = 0; r < 8; ++r) {
    int rg0 = mrow0 + M0a + r + 8 * hi;
    int rg1 = mrow0 + M0b + r + 8 * hi;
    A0[(size_t)rg0 * H + col] = (f16)(c0[r] + bias);
    A0[(size_t)rg1 * H + col] = (f16)(c1[r] + bias);
  }
}

// ---------------------------------------------------------------------------
// Per-step layer 0: h0n = tanh(A0_t + h0c @ W_hh0^T)
// M=B=64 (one block tile), grid = H/64 blocks
// ---------------------------------------------------------------------------
__global__ __launch_bounds__(256)
void step_layer0(const f16* __restrict__ h0c, const f16* __restrict__ Whh0,
                 const f16* __restrict__ A0t, f16* __restrict__ h0n) {
  __shared__ f16 As[2][64][LDT];
  __shared__ f16 Ws[2][64][LDT];
  int tid = threadIdx.x, lane = tid & 31, w = tid >> 5;
  int N0 = (w >> 1) * 16, M0a = (w & 1) * 32, M0b = M0a + 16;
  int ncol0 = blockIdx.x * 64;
  v8f c0 = {}, c1 = {};

  int row = tid >> 2, cc = tid & 3;
  const f16* aChunk = h0c + (size_t)row * H + cc * 8;
  const f16* wChunk = Whh0 + (size_t)(ncol0 + row) * H + cc * 8;

  gemm_k_loop(As, Ws, aChunk, wChunk, H / KT, tid, lane, N0, M0a, M0b, c0, c1);

  int mr = lane & 15, hi = lane >> 4;
  int col = ncol0 + N0 + mr;
#pragma unroll
  for (int r = 0; r < 8; ++r) {
    int b0 = M0a + r + 8 * hi;
    int b1 = M0b + r + 8 * hi;
    h0n[(size_t)b0 * H + col] = (f16)tanhf(c0[r] + (float)A0t[(size_t)b0 * H + col]);
    h0n[(size_t)b1 * H + col] = (f16)tanhf(c1[r] + (float)A0t[(size_t)b1 * H + col]);
  }
}

// ---------------------------------------------------------------------------
// Per-step layer 1 (fused): h1n = tanh(h0n @ W_ih1^T + b_ih1 + h1c @ W_hh1^T)
// also scatters h1n into the output sequence [B*T, H] at row b*T + t.
// ---------------------------------------------------------------------------
__global__ __launch_bounds__(256)
void step_layer1(const f16* __restrict__ h0n, const f16* __restrict__ Wih1,
                 const f16* __restrict__ h1c, const f16* __restrict__ Whh1,
                 const float* __restrict__ bih1, f16* __restrict__ h1n,
                 f16* __restrict__ outseq, int t) {
  __shared__ f16 As[2][64][LDT];
  __shared__ f16 Ws[2][64][LDT];
  int tid = threadIdx.x, lane = tid & 31, w = tid >> 5;
  int N0 = (w >> 1) * 16, M0a = (w & 1) * 32, M0b = M0a + 16;
  int ncol0 = blockIdx.x * 64;
  v8f c0 = {}, c1 = {};

  int row = tid >> 2, cc = tid & 3;
  // input-path GEMM
  gemm_k_loop(As, Ws,
              h0n + (size_t)row * H + cc * 8,
              Wih1 + (size_t)(ncol0 + row) * H + cc * 8,
              H / KT, tid, lane, N0, M0a, M0b, c0, c1);
  // recurrent-path GEMM, same accumulators
  gemm_k_loop(As, Ws,
              h1c + (size_t)row * H + cc * 8,
              Whh1 + (size_t)(ncol0 + row) * H + cc * 8,
              H / KT, tid, lane, N0, M0a, M0b, c0, c1);

  int mr = lane & 15, hi = lane >> 4;
  int col = ncol0 + N0 + mr;
  float bias = bih1[col];
#pragma unroll
  for (int r = 0; r < 8; ++r) {
    int b0 = M0a + r + 8 * hi;
    int b1 = M0b + r + 8 * hi;
    f16 v0 = (f16)tanhf(c0[r] + bias);
    f16 v1 = (f16)tanhf(c1[r] + bias);
    h1n[(size_t)b0 * H + col] = v0;
    h1n[(size_t)b1 * H + col] = v1;
    outseq[((size_t)b0 * T + t) * H + col] = v0;
    outseq[((size_t)b1 * T + t) * H + col] = v1;
  }
}

// ---------------------------------------------------------------------------
// GEMM 3 (parallel): logits[m][v] = outseq[m] . fc_W[v] + fc_b[v]  (f32 out)
// m = b*T + t ; grid = (V/64, B*T/64)
// ---------------------------------------------------------------------------
__global__ __launch_bounds__(256)
void out_projection(const f16* __restrict__ outseq, const f16* __restrict__ fcW,
                    const float* __restrict__ fcb, float* __restrict__ out) {
  __shared__ f16 As[2][64][LDT];
  __shared__ f16 Ws[2][64][LDT];
  int tid = threadIdx.x, lane = tid & 31, w = tid >> 5;
  int N0 = (w >> 1) * 16, M0a = (w & 1) * 32, M0b = M0a + 16;
  int mrow0 = blockIdx.y * 64, ncol0 = blockIdx.x * 64;
  v8f c0 = {}, c1 = {};

  int row = tid >> 2, cc = tid & 3;
  gemm_k_loop(As, Ws,
              outseq + (size_t)(mrow0 + row) * H + cc * 8,
              fcW + (size_t)(ncol0 + row) * H + cc * 8,
              H / KT, tid, lane, N0, M0a, M0b, c0, c1);

  int mr = lane & 15, hi = lane >> 4;
  int col = ncol0 + N0 + mr;
  float bias = fcb[col];
#pragma unroll
  for (int r = 0; r < 8; ++r) {
    int rg0 = mrow0 + M0a + r + 8 * hi;
    int rg1 = mrow0 + M0b + r + 8 * hi;
    out[(size_t)rg0 * V + col] = c0[r] + bias;
    out[(size_t)rg1 * V + col] = c1[r] + bias;
  }
}

// ---------------------------------------------------------------------------
// Host side
// ---------------------------------------------------------------------------
extern "C" void kernel_launch(void* const* d_in, const int* in_sizes, int n_in,
                              void* d_out, int out_size, void* d_ws, size_t ws_size,
                              hipStream_t stream) {
  (void)in_sizes; (void)n_in; (void)out_size; (void)ws_size;

  const int*   x      = (const int*)d_in[0];
  const float* embed  = (const float*)d_in[1];
  const float* W_ih0  = (const float*)d_in[2];
  const float* b_ih0  = (const float*)d_in[3];
  const float* W_hh0  = (const float*)d_in[4];
  const float* W_ih1  = (const float*)d_in[5];
  const float* b_ih1  = (const float*)d_in[6];
  const float* W_hh1  = (const float*)d_in[7];
  const float* fc_W   = (const float*)d_in[8];
  const float* fc_b   = (const float*)d_in[9];
  float* out = (float*)d_out;

  // Carve workspace (256B-aligned chunks), ~278 MB total.
  char* p = (char*)d_ws;
  auto take = [&](size_t bytes) -> char* {
    char* r = p;
    p += (bytes + 255) & ~(size_t)255;
    return r;
  };
  f16* emb_h  = (f16*)take((size_t)V * E * 2);
  f16* wih0_h = (f16*)take((size_t)H * E * 2);
  f16* whh0_h = (f16*)take((size_t)H * H * 2);
  f16* wih1_h = (f16*)take((size_t)H * H * 2);
  f16* whh1_h = (f16*)take((size_t)H * H * 2);
  f16* fcw_h  = (f16*)take((size_t)V * H * 2);
  f16* h0b[2] = {(f16*)take((size_t)B * H * 2), (f16*)take((size_t)B * H * 2)};
  f16* h1b[2] = {(f16*)take((size_t)B * H * 2), (f16*)take((size_t)B * H * 2)};
  f16* A0     = (f16*)take((size_t)T * B * H * 2);   // pre-activations, [T*B, H]
  f16* outseq = (f16*)take((size_t)B * T * H * 2);   // h1 sequence, row = b*T + t

  auto cvt = [&](const float* s, f16* d, int n) {
    int blocks = (n + 1023) / 1024;
    cvt_f32_to_f16<<<blocks, 256, 0, stream>>>(s, d, n);
  };
  cvt(embed, emb_h,  V * E);
  cvt(W_ih0, wih0_h, H * E);
  cvt(W_hh0, whh0_h, H * H);
  cvt(W_ih1, wih1_h, H * H);
  cvt(W_hh1, whh1_h, H * H);
  cvt(fc_W,  fcw_h,  V * H);

  // Initial hidden states = 0 (buffer 0 is "current" at t=0).
  zero_f16<<<(B * H + 255) / 256, 256, 0, stream>>>(h0b[0], B * H);
  zero_f16<<<(B * H + 255) / 256, 256, 0, stream>>>(h1b[0], B * H);

  // Hoisted input GEMM for layer 0 over the full sequence.
  embed_input_gemm<<<dim3(H / 64, (B * T) / 64), 256, 0, stream>>>(
      x, emb_h, wih0_h, b_ih0, A0);

  // Sequential recurrence: 2 fused WMMA kernels per timestep, ping-pong state.
  for (int t = 0; t < T; ++t) {
    int cur = t & 1, nxt = cur ^ 1;
    step_layer0<<<dim3(H / 64), 256, 0, stream>>>(
        h0b[cur], whh0_h, A0 + (size_t)t * B * H, h0b[nxt]);
    step_layer1<<<dim3(H / 64), 256, 0, stream>>>(
        h0b[nxt], wih1_h, h1b[cur], whh1_h, b_ih1, h1b[nxt], outseq, t);
  }
  // After T (even) steps, final states live in buffer 0.

  // Hoisted output projection over the full sequence -> f32 logits.
  out_projection<<<dim3(V / 64, (B * T) / 64), 256, 0, stream>>>(
      outseq, fcw_h, fc_b, out);

  // Final h0, h1 -> f32 tail of d_out.
  tail_states<<<(B * H + 255) / 256, 256, 0, stream>>>(h0b[0], h1b[0], out);
}
